// NeRFMoE_81990925680841
// MI455X (gfx1250) — matrix operand order repaired and checked by reading, not compiled, not run on hardware
//
#include <hip/hip_runtime.h>
#include <math.h>

#define NPTS   32768
#define DMODEL 256
#define NEXP   8
#define LEXP   4

typedef __attribute__((ext_vector_type(16))) _Float16 v16h;
typedef __attribute__((ext_vector_type(8)))  float    v8f;
typedef __attribute__((ext_vector_type(4)))  unsigned su4;
typedef __attribute__((ext_vector_type(8)))  unsigned su8;

__device__ __forceinline__ v8f wmma_step(v16h a, v16h b, v8f c) {
  // D = A(16x32 f16) * B(32x16 f16) + C(16x16 f32)
  return __builtin_amdgcn_wmma_f32_16x16x32_f16(false, a, false, b, (short)0, c, false, false);
}

// Load an A fragment (16x32, f16) for this lane from a row-major f16 buffer.
// ISA layout: lanes0-15 v0..3:K=2v,2v+1 v4..7:K=2(v-4)+16..; lanes16-31 +8.
__device__ __forceinline__ v16h load_a32(const _Float16* p, int g) {
  union { v16h v; unsigned u[8]; } r;
  const unsigned* d = (const unsigned*)p;
#pragma unroll
  for (int vv = 0; vv < 8; ++vv) {
    int kp = (vv < 4) ? (2 * vv + 8 * g) : (2 * (vv - 4) + 16 + 8 * g);
    r.u[vv] = d[kp >> 1];
  }
  return r.v;
}

// ---- Tensor Data Mover: contiguous 16KB (8192 x f16) global -> LDS -------------
// D# per CDNA5 ISA: 2-group form (tensor <= 2D, VADDR2/3=NULL)
__device__ __forceinline__ void tdm_load_16kb(const _Float16* gsrc, unsigned lds_byte_addr) {
  unsigned long long ga = (unsigned long long)(uintptr_t)gsrc;
  su4 g0;
  g0[0] = 1u;                                   // count=1, user descriptor
  g0[1] = lds_byte_addr;                        // lds_addr (bytes)
  g0[2] = (unsigned)(ga & 0xFFFFFFFFu);         // global_addr[31:0]
  g0[3] = (unsigned)((ga >> 32) & 0x01FFFFFFu)  // global_addr[56:32]
          | (2u << 30);                         // type=2 ("image")
  su8 g1;
  g1[0] = 1u << 16;          // workgroup_mask=0 | data_size=1 (2 bytes)
  g1[1] = 0x2000u << 16;     // tensor_dim0[15:0]=8192 (bits 63:48)
  g1[2] = 1u << 16;          // tensor_dim0[31:16]=0 | tensor_dim1[15:0]=1
  g1[3] = 0x2000u << 16;     // tensor_dim1[31:16]=0 | tile_dim0=8192
  g1[4] = 1u;                // tile_dim1=1 | tile_dim2=0
  g1[5] = 8192u;             // tensor_dim0_stride[31:0]
  g1[6] = 0u;                // stride hi / tensor_dim1_stride lo
  g1[7] = 0u;                // tensor_dim1_stride hi
  asm volatile("tensor_load_to_lds %0, %1" :: "s"(g0), "s"(g1) : "memory");
}

// ---------------- weight packing: f32 [K,Nout] -> f16 B-fragments ----------------
__global__ void k_pack(const float* __restrict__ src, _Float16* __restrict__ dst,
                       int K, int Nout, int KT, int NT,
                       long long srcStride, long long dstStride) {
  src += (long long)blockIdx.y * srcStride;
  dst += (long long)blockIdx.y * dstStride;
  int tid = blockIdx.x * blockDim.x + threadIdx.x;
  int total = KT * NT * 512;
  if (tid >= total) return;
  int frag = tid >> 9;
  int w    = tid & 511;
  int lane = w >> 4;
  int i16  = w & 15;
  int v    = i16 >> 1;
  int sel  = i16 & 1;
  int kt = frag / NT, nt = frag % NT;
  int g = lane >> 4, nl = lane & 15;
  int k = kt * 32 + ((2 * v + sel) & 7) + 8 * g + 16 * (v >> 2);
  int n = nt * 16 + nl;
  float val = (k < K) ? src[(long long)k * Nout + n] : 0.0f;
  dst[tid] = (_Float16)val;
}

// ---------------- positional encodings (f16, padded) ----------------
__global__ void k_posenc(const float* __restrict__ xyz, const float* __restrict__ dirs,
                         _Float16* __restrict__ peX, _Float16* __restrict__ peD) {
  int i = blockIdx.x * blockDim.x + threadIdx.x;
  if (i >= NPTS) return;
  {
    float x0 = xyz[3 * i], x1 = xyz[3 * i + 1], x2 = xyz[3 * i + 2];
    _Float16* o = peX + (long long)i * 64;
    o[0] = (_Float16)x0; o[1] = (_Float16)x1; o[2] = (_Float16)x2;
    float f = 1.0f;
    for (int q = 0; q < 10; ++q) {
      o[3 + 6 * q + 0] = (_Float16)sinf(x0 * f);
      o[3 + 6 * q + 1] = (_Float16)sinf(x1 * f);
      o[3 + 6 * q + 2] = (_Float16)sinf(x2 * f);
      o[3 + 6 * q + 3] = (_Float16)cosf(x0 * f);
      o[3 + 6 * q + 4] = (_Float16)cosf(x1 * f);
      o[3 + 6 * q + 5] = (_Float16)cosf(x2 * f);
      f *= 2.0f;
    }
    o[63] = (_Float16)0.0f;
  }
  {
    float x0 = dirs[3 * i], x1 = dirs[3 * i + 1], x2 = dirs[3 * i + 2];
    _Float16* o = peD + (long long)i * 32;
    o[0] = (_Float16)x0; o[1] = (_Float16)x1; o[2] = (_Float16)x2;
    float f = 1.0f;
    for (int q = 0; q < 4; ++q) {
      o[3 + 6 * q + 0] = (_Float16)sinf(x0 * f);
      o[3 + 6 * q + 1] = (_Float16)sinf(x1 * f);
      o[3 + 6 * q + 2] = (_Float16)sinf(x2 * f);
      o[3 + 6 * q + 3] = (_Float16)cosf(x0 * f);
      o[3 + 6 * q + 4] = (_Float16)cosf(x1 * f);
      o[3 + 6 * q + 5] = (_Float16)cosf(x2 * f);
      f *= 2.0f;
    }
    for (int q = 27; q < 32; ++q) o[q] = (_Float16)0.0f;
  }
}

// ---------------- gate MLP + top-1 routing (VALU, fp32) ----------------
__global__ void k_gate(const float* __restrict__ xyz,
                       const float* __restrict__ gw0, const float* __restrict__ gb0,
                       const float* __restrict__ gw1, const float* __restrict__ gb1,
                       int* __restrict__ idx, float* __restrict__ top1) {
  int i = blockIdx.x * blockDim.x + threadIdx.x;
  if (i >= NPTS) return;
  float pe[63];
  float x0 = xyz[3 * i], x1 = xyz[3 * i + 1], x2 = xyz[3 * i + 2];
  pe[0] = x0; pe[1] = x1; pe[2] = x2;
  float f = 1.0f;
  for (int q = 0; q < 10; ++q) {
    pe[3 + 6 * q + 0] = sinf(x0 * f); pe[3 + 6 * q + 1] = sinf(x1 * f);
    pe[3 + 6 * q + 2] = sinf(x2 * f); pe[3 + 6 * q + 3] = cosf(x0 * f);
    pe[3 + 6 * q + 4] = cosf(x1 * f); pe[3 + 6 * q + 5] = cosf(x2 * f);
    f *= 2.0f;
  }
  float lg[NEXP];
  for (int e = 0; e < NEXP; ++e) lg[e] = gb1[e];
  for (int n = 0; n < 128; ++n) {
    float hn = gb0[n];
    for (int k = 0; k < 63; ++k) hn += pe[k] * gw0[k * 128 + n];
    hn = fmaxf(hn, 0.0f);
    for (int e = 0; e < NEXP; ++e) lg[e] += hn * gw1[n * NEXP + e];
  }
  float mx = lg[0]; int am = 0;
  for (int e = 1; e < NEXP; ++e) if (lg[e] > mx) { mx = lg[e]; am = e; }
  float s = 0.0f;
  for (int e = 0; e < NEXP; ++e) s += expf(lg[e] - mx);
  idx[i] = am;
  top1[i] = 1.0f / s;  // softmax value at the argmax
}

// ---------------- routing: counting sort into 16-aligned segments ----------------
__global__ void k_zero(int* __restrict__ perm, int np, int* __restrict__ counts) {
  int i = blockIdx.x * blockDim.x + threadIdx.x;
  if (i < np) perm[i] = -1;
  if (i < NEXP) counts[i] = 0;
}

__global__ void k_count(const int* __restrict__ idx, int* __restrict__ counts) {
  int i = blockIdx.x * blockDim.x + threadIdx.x;
  if (i < NPTS) atomicAdd(&counts[idx[i]], 1);
}

__global__ void k_offsets(const int* __restrict__ counts, int* __restrict__ segOff,
                          int* __restrict__ cursor) {
  if (threadIdx.x == 0 && blockIdx.x == 0) {
    int o = 0;
    for (int e = 0; e < NEXP; ++e) {
      segOff[e] = o;
      cursor[e] = o;
      o += (counts[e] + 15) & ~15;
    }
    segOff[NEXP] = o;
  }
}

__global__ void k_scatter(const int* __restrict__ idx, int* __restrict__ cursor,
                          int* __restrict__ perm) {
  int i = blockIdx.x * blockDim.x + threadIdx.x;
  if (i < NPTS) {
    int p = atomicAdd(&cursor[idx[i]], 1);
    perm[p] = i;
  }
}

// ---------------- generic WMMA GEMM (KT, strideK compile-time) ----------------
// block = 128 threads (4 waves); wave w computes 16 rows x cols [w*64, w*64+64)
template <int KT, int SK>
__global__ void k_gemm_nt16(const _Float16* __restrict__ X,
                            const _Float16* __restrict__ Wp,
                            const float* __restrict__ bias,
                            _Float16* __restrict__ Y, int doRelu) {
  const int mt   = blockIdx.x;
  const int lane = threadIdx.x & 31;
  const int wave = threadIdx.x >> 5;
  const int g    = lane >> 4;
  const int nl   = lane & 15;
  const int row  = mt * 16 + nl;
  const _Float16* arow  = X + (long long)row * SK;
  const _Float16* wlane = Wp + (long long)wave * 4 * 512 + lane * 16;

  v8f acc[4] = {};
#pragma unroll
  for (int kt = 0; kt < KT; ++kt) {
    v16h a = load_a32(arow + kt * 32, g);
    if (kt + 1 < KT) __builtin_prefetch(wlane + (kt + 1) * 16 * 512, 0, 0);
#pragma unroll
    for (int j = 0; j < 4; ++j) {
      v16h b = *(const v16h*)(wlane + (kt * 16 + j) * 512);
      acc[j] = wmma_step(a, b, acc[j]);
    }
  }
#pragma unroll
  for (int j = 0; j < 4; ++j) {
    int n = (wave * 4 + j) * 16 + nl;
    float bs = bias[n];
#pragma unroll
    for (int vv = 0; vv < 8; ++vv) {
      int m = vv + 8 * g;  // C/D: M = vgpr + 8*(lane>=16), N = lane&15
      float val = acc[j][vv] + bs;
      if (doRelu) val = fmaxf(val, 0.0f);
      Y[(long long)(mt * 16 + m) * DMODEL + n] = (_Float16)val;
    }
  }
}

// ---------------- expert MLP (4 layers, skip at layer 1) over routed tiles ----------------
// block = 128 threads (4 waves), one 16-row permuted tile. B fragments for each
// k-slice (16 frags = 16KB) are staged into LDS by the Tensor Data Mover,
// double-buffered, so the VMEM pipe only sees the activation gather.
__global__ void k_experts(const _Float16* __restrict__ h,
                          const _Float16* __restrict__ Wp,   // [L*E][128 frag][512]
                          const float* __restrict__ eb,      // [L,E,256]
                          const int* __restrict__ perm,
                          const int* __restrict__ segOff,    // [E+1], 16-aligned
                          const float* __restrict__ top1,
                          _Float16* __restrict__ y) {
  __shared__ _Float16 bufH[16 * DMODEL];
  __shared__ _Float16 buf0[16 * DMODEL];
  __shared__ _Float16 buf1[16 * DMODEL];
  __shared__ __align__(32) _Float16 wbuf[2][8192];  // TDM double buffer (2 x 16KB)
  __shared__ int   pr[16];
  __shared__ float tt[16];

  const int tile = blockIdx.x;
  if (tile * 16 >= segOff[NEXP]) return;
  int e = 0;
#pragma unroll
  for (int q = 1; q < NEXP; ++q)
    if (tile * 16 >= segOff[q]) e = q;

  if (threadIdx.x < 16) {
    int p = perm[tile * 16 + threadIdx.x];
    pr[threadIdx.x] = p;
    tt[threadIdx.x] = (p >= 0) ? top1[p] : 0.0f;
  }
  __syncthreads();

  // cooperative gather of the 16x256 h-tile into LDS (zeros for pad rows)
  {
    int r = threadIdx.x >> 3;           // 16 rows, 8 threads per row
    int c = (threadIdx.x & 7) * 32;     // 32 halfs per thread
    int src = pr[r];
    const uint4* gp = (const uint4*)(h + (long long)(src < 0 ? 0 : src) * DMODEL + c);
#pragma unroll
    for (int q = 0; q < 4; ++q) {
      uint4 val = make_uint4(0u, 0u, 0u, 0u);
      if (src >= 0) val = gp[q];
      *(uint4*)&bufH[r * DMODEL + c + q * 8] = val;
    }
  }
  __syncthreads();

  const int lane = threadIdx.x & 31;
  const int wave = threadIdx.x >> 5;
  const int g = lane >> 4, nl = lane & 15;

  const _Float16* cur = bufH;
  _Float16* nxt = buf0;

  for (int l = 0; l < LEXP; ++l) {
    const _Float16* wbase = Wp + (long long)(l * NEXP + e) * 128 * 512;
    if (wave == 0)
      tdm_load_16kb(wbase, (unsigned)(uintptr_t)&wbuf[0][0]);  // prime kt=0
    v8f acc[4] = {};
#pragma unroll
    for (int kt = 0; kt < 8; ++kt) {
      if (wave == 0) {
        if (kt < 7) {
          tdm_load_16kb(wbase + (kt + 1) * 8192,
                        (unsigned)(uintptr_t)&wbuf[(kt + 1) & 1][0]);
          __builtin_amdgcn_s_wait_tensorcnt(1);  // slice kt landed (in-order)
        } else {
          __builtin_amdgcn_s_wait_tensorcnt(0);
        }
      }
      __syncthreads();  // release consumers: wbuf[kt&1] valid
      v16h a = load_a32(cur + nl * DMODEL + kt * 32, g);
      const _Float16* wl = &wbuf[kt & 1][0] + (long long)wave * 4 * 512 + lane * 16;
#pragma unroll
      for (int j = 0; j < 4; ++j) {
        v16h b = *(const v16h*)(wl + j * 512);
        acc[j] = wmma_step(a, b, acc[j]);
      }
      __syncthreads();  // all waves done with wbuf[kt&1] before it is refilled
    }
    const float* bl = eb + (long long)(l * NEXP + e) * DMODEL;
#pragma unroll
    for (int j = 0; j < 4; ++j) {
      int n = (wave * 4 + j) * 16 + nl;
      float bs = bl[n];
#pragma unroll
      for (int vv = 0; vv < 8; ++vv) {
        int m = vv + 8 * g;
        float val = acc[j][vv] + bs;
        if (l == 1) val += (float)bufH[m * DMODEL + n];  // skip: + h
        if (l < LEXP - 1) val = fmaxf(val, 0.0f);        // relu except last
        if (l == LEXP - 1) {
          val *= tt[m];                                  // top-1 combine weight
          int src = pr[m];
          if (src >= 0) y[(long long)src * DMODEL + n] = (_Float16)val;
        } else {
          nxt[m * DMODEL + n] = (_Float16)val;
        }
      }
    }
    __syncthreads();
    cur = nxt;
    nxt = (cur == buf0) ? buf1 : buf0;
  }
}

// ---------------- color head (WMMA, K=288) + sigma head + activations ----------------
// block = 64 threads (2 waves) per 16-row tile
__global__ void k_final(const _Float16* __restrict__ y, const _Float16* __restrict__ peD,
                        const _Float16* __restrict__ cw0p, const float* __restrict__ cb0,
                        const float* __restrict__ cw1, const float* __restrict__ cb1,
                        const float* __restrict__ sw, const float* __restrict__ sb,
                        float* __restrict__ out) {
  __shared__ _Float16 ch[16 * 128];
  const int mt   = blockIdx.x;
  const int lane = threadIdx.x & 31;
  const int wave = threadIdx.x >> 5;  // 0..1
  const int g = lane >> 4, nl = lane & 15;
  const int row = mt * 16 + nl;
  const _Float16* wlane = cw0p + (long long)wave * 4 * 512 + lane * 16;

  v8f acc[4] = {};
#pragma unroll
  for (int kt = 0; kt < 9; ++kt) {
    v16h a = (kt < 8) ? load_a32(y + (long long)row * DMODEL + kt * 32, g)
                      : load_a32(peD + (long long)row * 32, g);
#pragma unroll
    for (int j = 0; j < 4; ++j) {
      v16h b = *(const v16h*)(wlane + (kt * 8 + j) * 512);
      acc[j] = wmma_step(a, b, acc[j]);
    }
  }
#pragma unroll
  for (int j = 0; j < 4; ++j) {
    int n = (wave * 4 + j) * 16 + nl;
    float bs = cb0[n];
#pragma unroll
    for (int vv = 0; vv < 8; ++vv) {
      int m = vv + 8 * g;
      float val = fmaxf(acc[j][vv] + bs, 0.0f);
      ch[m * 128 + n] = (_Float16)val;
    }
  }
  __syncthreads();

  if (threadIdx.x < 16) {
    int i = mt * 16 + threadIdx.x;
    float r0 = cb1[0], r1 = cb1[1], r2 = cb1[2];
    for (int n = 0; n < 128; ++n) {
      float hv = (float)ch[threadIdx.x * 128 + n];
      r0 += hv * cw1[n * 3 + 0];
      r1 += hv * cw1[n * 3 + 1];
      r2 += hv * cw1[n * 3 + 2];
    }
    float sg = sb[0];
    for (int k = 0; k < DMODEL; ++k) sg += (float)y[(long long)i * DMODEL + k] * sw[k];
    out[(long long)i * 4 + 0] = 1.0f / (1.0f + expf(-r0));
    out[(long long)i * 4 + 1] = 1.0f / (1.0f + expf(-r1));
    out[(long long)i * 4 + 2] = 1.0f / (1.0f + expf(-r2));
    out[(long long)i * 4 + 3] = (sg > 20.0f) ? sg : log1pf(expf(sg));
  }
}

// ======================================================================
extern "C" void kernel_launch(void* const* d_in, const int* in_sizes, int n_in,
                              void* d_out, int out_size, void* d_ws, size_t ws_size,
                              hipStream_t stream) {
  (void)in_sizes; (void)n_in; (void)out_size; (void)ws_size;
  const float* xyz      = (const float*)d_in[0];
  const float* dirs     = (const float*)d_in[1];
  const float* head_w0  = (const float*)d_in[2];
  const float* head_b0  = (const float*)d_in[3];
  const float* head_w1  = (const float*)d_in[4];
  const float* head_b1  = (const float*)d_in[5];
  const float* gate_w0  = (const float*)d_in[6];
  const float* gate_b0  = (const float*)d_in[7];
  const float* gate_w1  = (const float*)d_in[8];
  const float* gate_b1  = (const float*)d_in[9];
  const float* expert_w = (const float*)d_in[10];
  const float* expert_b = (const float*)d_in[11];
  const float* sigma_w  = (const float*)d_in[12];
  const float* sigma_b  = (const float*)d_in[13];
  const float* color_w0 = (const float*)d_in[14];
  const float* color_b0 = (const float*)d_in[15];
  const float* color_w1 = (const float*)d_in[16];
  const float* color_b1 = (const float*)d_in[17];
  float* out = (float*)d_out;

  // workspace carve-up (256B aligned)
  char* ws = (char*)d_ws;
  size_t off = 0;
  auto carve = [&](size_t bytes) -> char* {
    char* p = ws + off;
    off = (off + bytes + 255) & ~(size_t)255;
    return p;
  };
  _Float16* w0p  = (_Float16*)carve((size_t)2 * 16 * 512 * 2);          // head_w0 packed
  _Float16* w1p  = (_Float16*)carve((size_t)8 * 16 * 512 * 2);          // head_w1 packed
  _Float16* ewp  = (_Float16*)carve((size_t)32 * 128 * 512 * 2);        // experts packed
  _Float16* cw0p = (_Float16*)carve((size_t)9 * 8 * 512 * 2);           // color_w0 packed
  _Float16* peX  = (_Float16*)carve((size_t)NPTS * 64 * 2);
  _Float16* peD  = (_Float16*)carve((size_t)NPTS * 32 * 2);
  _Float16* t0   = (_Float16*)carve((size_t)NPTS * DMODEL * 2);         // head hidden, reused as y
  _Float16* hbuf = (_Float16*)carve((size_t)NPTS * DMODEL * 2);         // head output h
  int*   idx    = (int*)carve((size_t)NPTS * 4);
  float* tp1    = (float*)carve((size_t)NPTS * 4);
  int*   perm   = (int*)carve((size_t)(NPTS + 256) * 4);
  int*   counts = (int*)carve(NEXP * 4);
  int*   segOff = (int*)carve((NEXP + 1) * 4);
  int*   cursor = (int*)carve(NEXP * 4);
  _Float16* ybuf = t0;  // safe reuse: t0 fully consumed before experts run

  // 1) pack weights to WMMA B-fragment layout (f16)
  k_pack<<<dim3(64, 1), 256, 0, stream>>>(head_w0, w0p, 63, 256, 2, 16, 0, 0);
  k_pack<<<dim3(256, 1), 256, 0, stream>>>(head_w1, w1p, 256, 256, 8, 16, 0, 0);
  k_pack<<<dim3(256, 32), 256, 0, stream>>>(expert_w, ewp, 256, 256, 8, 16,
                                            (long long)DMODEL * DMODEL,
                                            (long long)8 * 16 * 512);
  k_pack<<<dim3(144, 1), 256, 0, stream>>>(color_w0, cw0p, 283, 128, 9, 8, 0, 0);

  // 2) positional encodings + gate/routing
  k_posenc<<<NPTS / 256, 256, 0, stream>>>(xyz, dirs, peX, peD);
  k_gate<<<NPTS / 256, 256, 0, stream>>>(xyz, gate_w0, gate_b0, gate_w1, gate_b1, idx, tp1);
  k_zero<<<(NPTS + 256 + 255) / 256, 256, 0, stream>>>(perm, NPTS + 256, counts);
  k_count<<<NPTS / 256, 256, 0, stream>>>(idx, counts);
  k_offsets<<<1, 1, 0, stream>>>(counts, segOff, cursor);
  k_scatter<<<NPTS / 256, 256, 0, stream>>>(idx, cursor, perm);

  // 3) head MLP: t0 = relu(peX @ W0 + b0); h = t0 @ W1 + b1
  k_gemm_nt16<2, 64><<<NPTS / 16, 128, 0, stream>>>(peX, w0p, head_b0, t0, 1);
  k_gemm_nt16<8, 256><<<NPTS / 16, 128, 0, stream>>>(t0, w1p, head_b1, hbuf, 0);

  // 4) routed expert MLPs (top-1 only): y = top1 * expert_e(h)
  int maxTiles = NPTS / 16 + NEXP;  // padded segments
  k_experts<<<maxTiles, 128, 0, stream>>>(hbuf, ewp, expert_b, perm, segOff, tp1, ybuf);

  // 5) color + sigma heads
  k_final<<<NPTS / 16, 64, 0, stream>>>(ybuf, peD, cw0p, color_b0, color_w1, color_b1,
                                        sigma_w, sigma_b, out);
}
